// RecurrentNeuralNetwork_3590592659720
// MI455X (gfx1250) — matrix operand order repaired and checked by reading, not compiled, not run on hardware
//
#include <hip/hip_runtime.h>
#include <hip/hip_fp16.h>
#include <stdint.h>

typedef __attribute__((ext_vector_type(16))) _Float16 v16h;
typedef __attribute__((ext_vector_type(8)))  _Float16 v8h;
typedef __attribute__((ext_vector_type(8)))  float    v8f;

#define BB 32
#define TT 64
#define NIN 64
#define NH 512
#define NO 16

#define SIGMA_NEU 0.05f
#define SIGMA_SYN 0.002f
#define NOISE_START 16

// ---- d_out layout (float elements): hidden_list, output_list, hidden, new_j
#define HL_OFF 0
#define OL_OFF (BB * TT * NH)
#define HF_OFF (OL_OFF + BB * TT * NO)
#define NJ_OFF (HF_OFF + BB * NH)

// ---- workspace layout (byte offsets)
#define WS_H    0
#define WS_A32  (WS_H   + BB * NH * 4)
#define WS_DJ   (WS_A32 + BB * NH * 4)
#define WS_A16  (WS_DJ  + BB * NH * 4)
#define WS_H16  (WS_A16 + BB * NH * 2)
#define WS_X16  (WS_H16 + BB * NH * 2)
#define WS_BPK  (WS_X16 + BB * NIN * 2 + 2048)           // packed B fragments (576x512 f16)
#define WS_WPK  (WS_BPK + (NH + NIN) * NH * 2)           // packed w_out^T fragments (512x16 f16)

#define KT_STEP 18   // 576 / 32
#define KT_OUT  16   // 512 / 32

#if defined(__has_builtin)
#  if __has_builtin(__builtin_amdgcn_global_load_async_to_lds_b32)
#    define HAVE_ASYNC_LDS 1
#  endif
#endif

// ---------------- RNG: counter hash -> approx standard normal ----------------
__device__ __forceinline__ uint32_t mix32(uint32_t x) {
  x ^= x >> 16; x *= 0x7feb352dU;
  x ^= x >> 15; x *= 0x846ca68bU;
  x ^= x >> 16;
  return x;
}
__device__ __forceinline__ float nrm(uint32_t s) {
  uint32_t h1 = mix32(s);
  uint32_t h2 = mix32(h1 ^ 0x9E3779B9U);
  const float k = 1.0f / 4294967296.0f;
  return ((float)h1 * k + (float)h2 * k - 1.0f) * 2.44948974f;
}

// ---------------- init: new_j = broadcast(w_hh) --------------------------------
__global__ void k_init_newj(float* __restrict__ newj, const float* __restrict__ whh) {
  size_t idx = (size_t)blockIdx.x * blockDim.x + threadIdx.x;
  if (idx < (size_t)BB * NH * NH) newj[idx] = whh[idx & (NH * NH - 1)];
}

// ---- init: packed WMMA B operands + h copy ------------------------------------
__global__ void k_init_misc(const float* __restrict__ whh, const float* __restrict__ win,
                            const float* __restrict__ wout, const float* __restrict__ h0,
                            _Float16* __restrict__ bpack, _Float16* __restrict__ wopack,
                            float* __restrict__ h) {
  int idx = blockIdx.x * blockDim.x + threadIdx.x;
  const int NBPK = 32 * KT_STEP * 32 * 16;     // 294912
  const int NWPK = KT_OUT * 32 * 16;           // 8192
  if (idx < NBPK) {
    int n = idx & 15;
    int lane = (idx >> 4) & 31;
    int kt = (idx >> 9) % KT_STEP;
    int Ntile = idx / (16 * 32 * KT_STEP);
    int K = kt * 32 + ((n & 8) ? 16 : 0) + ((lane & 16) ? 8 : 0) + (n & 7);
    int Nc = Ntile * 16 + (lane & 15);
    float v = (K < NH) ? (whh[Nc * NH + K] - whh[K * NH + Nc])
                       : win[Nc * NIN + (K - NH)];
    bpack[idx] = (_Float16)v;
  } else if (idx < NBPK + NWPK) {
    int p = idx - NBPK;
    int n = p & 15;
    int lane = (p >> 4) & 31;
    int kt = p >> 9;
    int K = kt * 32 + ((n & 8) ? 16 : 0) + ((lane & 16) ? 8 : 0) + (n & 7);
    int O = lane & 15;
    wopack[p] = (_Float16)wout[O * NH + K];
  } else if (idx < NBPK + NWPK + BB * NH) {
    int k = idx - NBPK - NWPK;
    h[k] = h0[k];
  }
}

// ---------------- per step: a = tanh(h), zero dj, stage x_t --------------------
__global__ void k_act(const float* __restrict__ h, float* __restrict__ a32,
                      _Float16* __restrict__ a16, float* __restrict__ dj,
                      const float* __restrict__ input, _Float16* __restrict__ x16,
                      int t) {
  int idx = blockIdx.x * blockDim.x + threadIdx.x;
  if (idx < BB * NH) {
    float av = tanhf(h[idx]);
    a32[idx] = av;
    a16[idx] = (_Float16)av;
    dj[idx] = 0.0f;
  }
  if (idx < BB * NIN) {
    int b = idx >> 6, k = idx & 63;
    x16[idx] = (_Float16)input[(b * TT + t) * NIN + k];
  }
}

// ---------------- per step: fused new_j stream, float4-vectorized --------------
// grid: 1024 blocks (32 batches x 32 i-chunks of 16 rows), 128 threads (4 j each)
__global__ void k_plastic(float* __restrict__ newj, const float* __restrict__ a32,
                          const float* __restrict__ beta, float* __restrict__ dj,
                          int t) {
  const int b  = blockIdx.x >> 5;
  const int ic = blockIdx.x & 31;
  const int i0 = ic * 16;
  const int j0 = threadIdx.x * 4;

  __shared__ float as_[16];
#if HAVE_ASYNC_LDS
  if (threadIdx.x < 16) {
    __builtin_amdgcn_global_load_async_to_lds_b32(
        (int*)(const_cast<float*>(a32) + b * NH + i0 + threadIdx.x),
        (int*)(&as_[threadIdx.x]), 0, 0);
  }
  asm volatile("s_wait_asynccnt 0" ::: "memory");
  __syncthreads();
#else
  if (threadIdx.x < 16) as_[threadIdx.x] = a32[b * NH + i0 + threadIdx.x];
  __syncthreads();
#endif

  const float4 aj = *(const float4*)(a32 + b * NH + j0);
  const float4 bj = *(const float4*)(beta + j0);
  const float cN0 = bj.x * SIGMA_SYN * sqrtf(bj.x), cH0 = -bj.x * aj.x;
  const float cN1 = bj.y * SIGMA_SYN * sqrtf(bj.y), cH1 = -bj.y * aj.y;
  const float cN2 = bj.z * SIGMA_SYN * sqrtf(bj.z), cH2 = -bj.z * aj.z;
  const float cN3 = bj.w * SIGMA_SYN * sqrtf(bj.w), cH3 = -bj.w * aj.w;

  float* base = newj + (size_t)b * NH * NH + j0;
  float d0 = 0.f, d1 = 0.f, d2 = 0.f, d3 = 0.f;
  const uint32_t stb = (uint32_t)t * 0x9E3779B9u ^ (uint32_t)b * 0x85ebca6bu;

  #pragma unroll 4
  for (int ii = 0; ii < 16; ++ii) {
    const int i = i0 + ii;
    float* p = base + (size_t)i * NH;
    if (ii < 12) __builtin_prefetch(base + (size_t)(i + 4) * NH, 0, 1);
    float4 v = *(float4*)p;
    const float ai = as_[ii];
    d0 = fmaf(ai, v.x, d0);
    d1 = fmaf(ai, v.y, d1);
    d2 = fmaf(ai, v.z, d2);
    d3 = fmaf(ai, v.w, d3);
    const uint32_t si = stb ^ (uint32_t)i * 0xc2b2ae35u;
    v.x = fmaf(cN0, nrm(si ^ (uint32_t)(j0 + 0) * 0x27d4eb2fu), fmaf(ai, cH0, v.x));
    v.y = fmaf(cN1, nrm(si ^ (uint32_t)(j0 + 1) * 0x27d4eb2fu), fmaf(ai, cH1, v.y));
    v.z = fmaf(cN2, nrm(si ^ (uint32_t)(j0 + 2) * 0x27d4eb2fu), fmaf(ai, cH2, v.z));
    v.w = fmaf(cN3, nrm(si ^ (uint32_t)(j0 + 3) * 0x27d4eb2fu), fmaf(ai, cH3, v.w));
    *(float4*)p = v;
  }
  atomicAdd(&dj[b * NH + j0 + 0], d0);
  atomicAdd(&dj[b * NH + j0 + 1], d1);
  atomicAdd(&dj[b * NH + j0 + 2], d2);
  atomicAdd(&dj[b * NH + j0 + 3], d3);
}

// ---------------- per step: WMMA GEMM [32x576]x[576x512] + h-update epilogue ----
__global__ void k_step_gemm(const float* __restrict__ bhh, const float* __restrict__ alpha,
                            const _Float16* __restrict__ a16, const _Float16* __restrict__ x16,
                            const _Float16* __restrict__ bpack, const float* __restrict__ dj,
                            float* __restrict__ h, _Float16* __restrict__ h16,
                            float* __restrict__ out_hl, int t) {
  const int lane  = threadIdx.x;
  const int Mtile = blockIdx.x & 1;
  const int Ntile = blockIdx.x >> 1;
  const int mrow  = lane & 15;
  const int hiA   = (lane & 16) ? 8 : 0;
  const int jcol  = Ntile * 16 + (lane & 15);
  const int brow  = Mtile * 16 + mrow;

  const _Float16* bp = bpack + (((size_t)Ntile * KT_STEP * 32) + lane) * 16;

  v8f c = {};
  #pragma unroll
  for (int kt = 0; kt < KT_STEP; ++kt) {
    const int kbase = kt * 32;
    union { v16h v; v8h h2[2]; } af, bf;
    const _Float16* arow = (kbase < NH) ? (a16 + brow * NH + kbase)
                                        : (x16 + brow * NIN + (kbase - NH));
    af.h2[0] = *(const v8h*)(arow + hiA);
    af.h2[1] = *(const v8h*)(arow + 16 + hiA);
    const _Float16* bpt = bp + (size_t)kt * 32 * 16;
    bf.h2[0] = *(const v8h*)(bpt);
    bf.h2[1] = *(const v8h*)(bpt + 8);
    c = __builtin_amdgcn_wmma_f32_16x16x32_f16(false, af.v, false, bf.v,
                                               (short)0, c, false, false);
  }

  const int mhalf = (lane >> 4) & 1;
  const float alj = alpha[jcol];
  const float bhj = bhh[jcol];
  const float sna = SIGMA_NEU * sqrtf(alj);
  #pragma unroll
  for (int r = 0; r < 8; ++r) {
    int bb = Mtile * 16 + r + mhalf * 8;
    float tmp = c[r] + bhj + dj[bb * NH + jcol];
    float hn = (1.0f - alj) * h[bb * NH + jcol] + alj * tmp;
    if (t >= NOISE_START) {
      uint32_t seed = (uint32_t)t * 0x9E3779B9u ^ (uint32_t)bb * 0x85ebca6bu
                    ^ (uint32_t)jcol * 0xc2b2ae35u ^ 0x51ed270bu;
      hn = fmaf(nrm(seed), sna, hn);
    }
    h[bb * NH + jcol] = hn;
    h16[bb * NH + jcol] = (_Float16)hn;
    out_hl[(bb * TT + t) * NH + jcol] = hn;
  }
}

// ---------------- per step: out = h @ w_out^T via WMMA --------------------------
__global__ void k_out_gemm(const _Float16* __restrict__ h16,
                           const _Float16* __restrict__ wopack,
                           float* __restrict__ out_ol, int t) {
  const int lane  = threadIdx.x;
  const int Mtile = blockIdx.x;
  const int mrow  = lane & 15;
  const int hiA   = (lane & 16) ? 8 : 0;
  const int ocol  = lane & 15;
  const int brow  = Mtile * 16 + mrow;

  v8f c = {};
  #pragma unroll
  for (int kt = 0; kt < KT_OUT; ++kt) {
    const int kbase = kt * 32;
    union { v16h v; v8h h2[2]; } af, bf;
    const _Float16* arow = h16 + brow * NH + kbase;
    af.h2[0] = *(const v8h*)(arow + hiA);
    af.h2[1] = *(const v8h*)(arow + 16 + hiA);
    const _Float16* bpt = wopack + ((size_t)kt * 32 + lane) * 16;
    bf.h2[0] = *(const v8h*)(bpt);
    bf.h2[1] = *(const v8h*)(bpt + 8);
    c = __builtin_amdgcn_wmma_f32_16x16x32_f16(false, af.v, false, bf.v,
                                               (short)0, c, false, false);
  }
  const int mhalf = (lane >> 4) & 1;
  #pragma unroll
  for (int r = 0; r < 8; ++r) {
    int bb = Mtile * 16 + r + mhalf * 8;
    out_ol[(bb * TT + t) * NO + ocol] = c[r];
  }
}

// ---------------- final hidden copy --------------------------------------------
__global__ void k_final(const float* __restrict__ h, float* __restrict__ hf) {
  int idx = blockIdx.x * blockDim.x + threadIdx.x;
  if (idx < BB * NH) hf[idx] = h[idx];
}

extern "C" void kernel_launch(void* const* d_in, const int* in_sizes, int n_in,
                              void* d_out, int out_size, void* d_ws, size_t ws_size,
                              hipStream_t stream) {
  (void)in_sizes; (void)n_in; (void)out_size; (void)ws_size;

  const float* input = (const float*)d_in[0];
  const float* h0    = (const float*)d_in[1];
  const float* win   = (const float*)d_in[2];
  const float* whh   = (const float*)d_in[3];
  const float* bhh   = (const float*)d_in[4];
  const float* wout  = (const float*)d_in[5];
  const float* alpha = (const float*)d_in[6];
  const float* beta  = (const float*)d_in[7];

  float* out    = (float*)d_out;
  float* out_hl = out + HL_OFF;
  float* out_ol = out + OL_OFF;
  float* out_hf = out + HF_OFF;
  float* newj   = out + NJ_OFF;

  char* ws = (char*)d_ws;
  float*    h    = (float*)(ws + WS_H);
  float*    a32  = (float*)(ws + WS_A32);
  float*    dj   = (float*)(ws + WS_DJ);
  _Float16* a16  = (_Float16*)(ws + WS_A16);
  _Float16* h16  = (_Float16*)(ws + WS_H16);
  _Float16* x16  = (_Float16*)(ws + WS_X16);
  _Float16* bpk  = (_Float16*)(ws + WS_BPK);
  _Float16* wpk  = (_Float16*)(ws + WS_WPK);

  k_init_newj<<<(BB * NH * NH + 255) / 256, 256, 0, stream>>>(newj, whh);
  {
    int total = 32 * KT_STEP * 32 * 16 + KT_OUT * 32 * 16 + BB * NH;
    k_init_misc<<<(total + 255) / 256, 256, 0, stream>>>(whh, win, wout, h0, bpk, wpk, h);
  }

  for (int t = 0; t < TT; ++t) {
    k_act<<<(BB * NH + 255) / 256, 256, 0, stream>>>(h, a32, a16, dj, input, x16, t);
    k_plastic<<<1024, 128, 0, stream>>>(newj, a32, beta, dj, t);
    k_step_gemm<<<64, 32, 0, stream>>>(bhh, alpha, a16, x16, bpk, dj, h, h16, out_hl, t);
    k_out_gemm<<<2, 32, 0, stream>>>(h16, wpk, out_ol, t);
  }

  k_final<<<(BB * NH + 255) / 256, 256, 0, stream>>>(h, out_hf);
}
